// DeeperGCN_TU_44555990729012
// MI455X (gfx1250) — matrix operand order repaired and checked by reading, not compiled, not run on hardware
//
#include <hip/hip_runtime.h>

#define NN 16384
#define EE 160000
#define HH 512
#define LL 7
#define GG 128
#define LN_EPS 1e-5f
#define MEPS   1e-7f

typedef __bf16 bf16;
typedef __attribute__((ext_vector_type(16))) __bf16 v16bf;
typedef __attribute__((ext_vector_type(8)))  __bf16 v8bf;
typedef __attribute__((ext_vector_type(8)))  float  v8f;

// ---------------------------------------------------------------- utilities
__global__ void fill_zero_kernel(float4* __restrict__ p, int n4) {
    int i = blockIdx.x * blockDim.x + threadIdx.x;
    if (i < n4) p[i] = make_float4(0.f, 0.f, 0.f, 0.f);
}

__global__ void copy_f32_kernel(const float* __restrict__ src, float* __restrict__ dst, int n) {
    int i = blockIdx.x * blockDim.x + threadIdx.x;
    if (i < n) dst[i] = src[i];
}

// h = x + vn_emb (broadcast over all nodes; vn[batch] == vn_emb at layer 0)
__global__ void init_h_kernel(const float* __restrict__ x, const float* __restrict__ vn_emb,
                              float* __restrict__ h, int total) {
    int i = blockIdx.x * blockDim.x + threadIdx.x;
    if (i < total) h[i] = x[i] + vn_emb[i & (HH - 1)];
}

__global__ void init_vn_kernel(const float* __restrict__ vn_emb, float* __restrict__ vn, int total) {
    int i = blockIdx.x * blockDim.x + threadIdx.x;
    if (i < total) vn[i] = vn_emb[i & (HH - 1)];
}

// bf16 transposed weight:  Wt[l][n][k] = W[l][k][n]
__global__ void conv_weights_kernel(const float* __restrict__ W, bf16* __restrict__ Wt, int total) {
    int i = blockIdx.x * blockDim.x + threadIdx.x;
    if (i >= total) return;
    int l = i >> 18;            // / (512*512)
    int rem = i & 262143;
    int n = rem >> 9;
    int k = rem & 511;
    Wt[i] = (bf16)W[l * 262144 + k * HH + n];
}

// ------------------------------------------------------- edge softmax aggr
// pass 1: msg = relu(h[src]) + eps ; segment max over dst (uint atomicMax,
// valid because all msgs > 0 so the f32 bit pattern is order-preserving)
__global__ void edge_max_kernel(const float* __restrict__ hin,
                                const int* __restrict__ src, const int* __restrict__ dst,
                                unsigned* __restrict__ mmax) {
    int gid = blockIdx.x * blockDim.x + threadIdx.x;
    if (gid >= EE * (HH / 4)) return;
    int e = gid >> 7;               // / 128
    int c = (gid & 127) << 2;
    int s = src[e], d = dst[e];
    float4 hv = *(const float4*)(hin + (size_t)s * HH + c);
    float m0 = fmaxf(hv.x, 0.f) + MEPS;
    float m1 = fmaxf(hv.y, 0.f) + MEPS;
    float m2 = fmaxf(hv.z, 0.f) + MEPS;
    float m3 = fmaxf(hv.w, 0.f) + MEPS;
    unsigned* mp = mmax + (size_t)d * HH + c;
    atomicMax(mp + 0, __float_as_uint(m0));
    atomicMax(mp + 1, __float_as_uint(m1));
    atomicMax(mp + 2, __float_as_uint(m2));
    atomicMax(mp + 3, __float_as_uint(m3));
}

// pass 2: ex = exp(msg - max); denom += ex ; wsum += msg*ex
__global__ void edge_sum_kernel(const float* __restrict__ hin,
                                const int* __restrict__ src, const int* __restrict__ dst,
                                const unsigned* __restrict__ mmax,
                                float* __restrict__ denom, float* __restrict__ wsum) {
    int gid = blockIdx.x * blockDim.x + threadIdx.x;
    if (gid >= EE * (HH / 4)) return;
    int e = gid >> 7;
    int c = (gid & 127) << 2;
    int s = src[e], d = dst[e];
    float4 hv = *(const float4*)(hin + (size_t)s * HH + c);
    const unsigned* mp = mmax + (size_t)d * HH + c;
    float m[4] = { fmaxf(hv.x, 0.f) + MEPS, fmaxf(hv.y, 0.f) + MEPS,
                   fmaxf(hv.z, 0.f) + MEPS, fmaxf(hv.w, 0.f) + MEPS };
    float* dp = denom + (size_t)d * HH + c;
    float* wp = wsum  + (size_t)d * HH + c;
    #pragma unroll
    for (int i = 0; i < 4; ++i) {
        float mx = __uint_as_float(mp[i]);
        float ex = __expf(m[i] - mx);
        atomicAdd(dp + i, ex);
        atomicAdd(wp + i, m[i] * ex);
    }
}

// t = hin + wsum / (denom > 0 ? denom : 1); store bf16 GEMM operand
__global__ void combine_kernel(const float* __restrict__ hin,
                               const float* __restrict__ denom, const float* __restrict__ wsum,
                               bf16* __restrict__ Abf, int total) {
    int i = blockIdx.x * blockDim.x + threadIdx.x;
    if (i >= total) return;
    float d = denom[i];
    d = (d > 0.f) ? d : 1.f;
    Abf[i] = (bf16)(hin[i] + wsum[i] / d);
}

// -------------------------------------------------------- bf16 WMMA GEMM
// C[rows,512] = A[rows,512](bf16) @ Bt^T + bias (+ resid), Bt is [n][k] bf16.
// One wave computes a 16x64 strip (4 N-tiles): A fragment reused 4x per k-step.
template<bool HAS_RESID>
__global__ __launch_bounds__(256) void gemm_wmma_kernel(
        const bf16* __restrict__ A, const bf16* __restrict__ Bt,
        const float* __restrict__ bias, const float* __restrict__ resid,
        float* __restrict__ C, int rows) {
    int tid  = threadIdx.x;
    int lane = tid & 31;
    int wave = blockIdx.x * 8 + (tid >> 5);     // 8 waves / block
    int tile_m  = wave >> 3;                    // / (512/64)
    int tile_ng = wave & 7;                     // N-group of 4 tiles (64 cols)
    if (tile_m * 16 >= rows) return;

    int half = lane >> 4;                       // 0: lanes 0-15, 1: lanes 16-31
    int l16  = lane & 15;
    const bf16* Aptr = A  + (size_t)(tile_m * 16 + l16) * HH + half * 16;
    const bf16* B0   = Bt + (size_t)(tile_ng * 64 + l16) * HH + half * 16;

    v8f acc0 = {}, acc1 = {}, acc2 = {}, acc3 = {};
    #pragma unroll 2
    for (int kk = 0; kk < HH; kk += 32) {
        // A 16x32 bf16 layout: v0-3 <- K[kk+half*16 .. +7], v4-7 <- K[kk+half*16+8 .. +7]
        v8bf a0 = *(const v8bf*)(Aptr + kk);
        v8bf a1 = *(const v8bf*)(Aptr + kk + 8);
        v16bf a;
        #pragma unroll
        for (int i = 0; i < 8; ++i) { a[i] = a0[i]; a[i + 8] = a1[i]; }
        // B 32x16 bf16 layout: lane holds N=l16, K contiguous (transposed storage)
        v16bf b0 = *(const v16bf*)(B0 + kk);
        v16bf b1 = *(const v16bf*)(B0 + (size_t)16 * HH + kk);
        v16bf b2 = *(const v16bf*)(B0 + (size_t)32 * HH + kk);
        v16bf b3 = *(const v16bf*)(B0 + (size_t)48 * HH + kk);
        acc0 = __builtin_amdgcn_wmma_f32_16x16x32_bf16(false, a, false, b0, (short)0, acc0, false, false);
        acc1 = __builtin_amdgcn_wmma_f32_16x16x32_bf16(false, a, false, b1, (short)0, acc1, false, false);
        acc2 = __builtin_amdgcn_wmma_f32_16x16x32_bf16(false, a, false, b2, (short)0, acc2, false, false);
        acc3 = __builtin_amdgcn_wmma_f32_16x16x32_bf16(false, a, false, b3, (short)0, acc3, false, false);
    }

    v8f accs[4] = { acc0, acc1, acc2, acc3 };
    #pragma unroll
    for (int t = 0; t < 4; ++t) {
        int bcol = tile_ng * 64 + t * 16 + l16;
        float bv = bias[bcol];
        #pragma unroll
        for (int r = 0; r < 8; ++r) {
            int row = tile_m * 16 + r + half * 8;
            size_t idx = (size_t)row * HH + bcol;
            float v = accs[t][r] + bv;
            if (HAS_RESID) v += resid[idx];
            C[idx] = v;
        }
    }
}

// ----------------------------------------------------------- LayerNorm
// one wave per row of length 512 (16 elements per lane)
__global__ __launch_bounds__(256) void ln_kernel(const float* __restrict__ in, float* __restrict__ out,
                                                 const float* __restrict__ g, const float* __restrict__ b,
                                                 int rows, int do_relu) {
    int wid  = (blockIdx.x * blockDim.x + threadIdx.x) >> 5;
    int lane = threadIdx.x & 31;
    if (wid >= rows) return;
    const float* r = in + (size_t)wid * HH + lane * 16;
    float v[16];
    #pragma unroll
    for (int i = 0; i < 4; ++i) {
        float4 t = *(const float4*)(r + i * 4);
        v[4 * i] = t.x; v[4 * i + 1] = t.y; v[4 * i + 2] = t.z; v[4 * i + 3] = t.w;
    }
    float s = 0.f, s2 = 0.f;
    #pragma unroll
    for (int i = 0; i < 16; ++i) { s += v[i]; s2 += v[i] * v[i]; }
    #pragma unroll
    for (int off = 16; off > 0; off >>= 1) {
        s  += __shfl_xor(s,  off, 32);
        s2 += __shfl_xor(s2, off, 32);
    }
    float mean = s * (1.f / HH);
    float var  = s2 * (1.f / HH) - mean * mean;
    float rs   = rsqrtf(var + LN_EPS);
    float* o = out + (size_t)wid * HH + lane * 16;
    #pragma unroll
    for (int i = 0; i < 16; ++i) {
        int col = lane * 16 + i;
        float y = (v[i] - mean) * rs * g[col] + b[col];
        if (do_relu) y = fmaxf(y, 0.f);
        o[i] = y;
    }
}

// ---------------------------------------------------- virtual-node pieces
__global__ void vn_scatter_kernel(const float* __restrict__ h2, const int* __restrict__ batch,
                                  float* __restrict__ vntmp) {
    int gid = blockIdx.x * blockDim.x + threadIdx.x;
    if (gid >= NN * (HH / 4)) return;
    int n = gid >> 7;
    int c = (gid & 127) << 2;
    int gg = batch[n];
    float4 hv = *(const float4*)(h2 + (size_t)n * HH + c);
    float* p = vntmp + (size_t)gg * HH + c;
    atomicAdd(p + 0, hv.x); atomicAdd(p + 1, hv.y);
    atomicAdd(p + 2, hv.z); atomicAdd(p + 3, hv.w);
}

__global__ __launch_bounds__(128) void vn_gemm_kernel(const float* __restrict__ X,
                                                      const float* __restrict__ W,
                                                      const float* __restrict__ bias,
                                                      float* __restrict__ Y) {
    int gidx = blockIdx.x;
    int j = blockIdx.y * 128 + threadIdx.x;
    const float* xr = X + (size_t)gidx * HH;
    float acc = bias[j];
    #pragma unroll 4
    for (int k = 0; k < HH; ++k) acc += xr[k] * W[(size_t)k * HH + j];
    Y[(size_t)gidx * HH + j] = acc;
}

__global__ void add_vn_kernel(float* __restrict__ h2, const float* __restrict__ vn,
                              const int* __restrict__ batch) {
    int gid = blockIdx.x * blockDim.x + threadIdx.x;
    if (gid >= NN * (HH / 4)) return;
    int n = gid >> 7;
    int c = (gid & 127) << 2;
    int gg = batch[n];
    float4 hv = *(float4*)(h2 + (size_t)n * HH + c);
    float4 vv = *(const float4*)(vn + (size_t)gg * HH + c);
    hv.x += vv.x; hv.y += vv.y; hv.z += vv.z; hv.w += vv.w;
    *(float4*)(h2 + (size_t)n * HH + c) = hv;
}

// ---------------------------------------------------------------- driver
extern "C" void kernel_launch(void* const* d_in, const int* in_sizes, int n_in,
                              void* d_out, int out_size, void* d_ws, size_t ws_size,
                              hipStream_t stream) {
    (void)in_sizes; (void)n_in; (void)out_size; (void)ws_size;
    const float* x        = (const float*)d_in[0];
    const int*   eidx     = (const int*)d_in[1];
    const int*   batch    = (const int*)d_in[2];
    const float* gcn_W    = (const float*)d_in[3];
    const float* gcn_b    = (const float*)d_in[4];
    const float* norm_g   = (const float*)d_in[5];
    const float* norm_b   = (const float*)d_in[6];
    const float* vn_emb   = (const float*)d_in[7];
    const float* vn_W1    = (const float*)d_in[8];
    const float* vn_b1    = (const float*)d_in[9];
    const float* vn_g     = (const float*)d_in[10];
    const float* vn_beta  = (const float*)d_in[11];
    const float* vn_W2    = (const float*)d_in[12];
    const float* vn_b2    = (const float*)d_in[13];
    const int* src = eidx;
    const int* dst = eidx + EE;

    char* w = (char*)d_ws;
    const size_t NH = (size_t)NN * HH;
    float* mmaxf = (float*)w;  w += NH * 4;     // keep mmax/denom/wsum contiguous
    float* denom = (float*)w;  w += NH * 4;
    float* wsum  = (float*)w;  w += NH * 4;
    float* hbuf  = (float*)w;  w += NH * 4;
    float* h2    = (float*)w;  w += NH * 4;
    bf16*  Abf   = (bf16*)w;   w += NH * 2;
    bf16*  Wt    = (bf16*)w;   w += (size_t)LL * HH * HH * 2;
    float* vnst  = (float*)w;  w += (size_t)GG * HH * 4;
    float* vntmp = (float*)w;  w += (size_t)GG * HH * 4;
    float* vnhid = (float*)w;  w += (size_t)GG * HH * 4;

    const int T = 256;
    const int nhBlk    = (int)((NH + T - 1) / T);
    const int edgeBlk  = (EE * (HH / 4) + T - 1) / T;
    const int nodeBlk  = (NN * (HH / 4) + T - 1) / T;
    const int zeroBlk  = (int)((3 * NH / 4 + T - 1) / T);
    const int gemmBlk  = (NN / 16) * (HH / 64) / 8;   // 1 wave = 16x64 strip, 8 waves/block
    const int lnBlkN   = (NN * 32 + T - 1) / T;
    const int lnBlkG   = (GG * 32 + T - 1) / T;
    const int ghBlk    = (GG * HH + T - 1) / T;
    const int wtBlk    = (LL * HH * HH + T - 1) / T;

    // one-time prep
    conv_weights_kernel<<<wtBlk, T, 0, stream>>>(gcn_W, Wt, LL * HH * HH);
    init_h_kernel<<<nhBlk, T, 0, stream>>>(x, vn_emb, hbuf, (int)NH);
    init_vn_kernel<<<ghBlk, T, 0, stream>>>(vn_emb, vnst, GG * HH);

    auto genconv = [&](const float* hin, int layer, const float* resid, float* outp) {
        fill_zero_kernel<<<zeroBlk, T, 0, stream>>>((float4*)mmaxf, (int)(3 * NH / 4));
        edge_max_kernel<<<edgeBlk, T, 0, stream>>>(hin, src, dst, (unsigned*)mmaxf);
        edge_sum_kernel<<<edgeBlk, T, 0, stream>>>(hin, src, dst, (const unsigned*)mmaxf,
                                                   denom, wsum);
        combine_kernel<<<nhBlk, T, 0, stream>>>(hin, denom, wsum, Abf, (int)NH);
        if (resid)
            gemm_wmma_kernel<true><<<gemmBlk, T, 0, stream>>>(
                Abf, Wt + (size_t)layer * HH * HH, gcn_b + (size_t)layer * HH, resid, outp, NN);
        else
            gemm_wmma_kernel<false><<<gemmBlk, T, 0, stream>>>(
                Abf, Wt + (size_t)layer * HH * HH, gcn_b + (size_t)layer * HH, nullptr, outp, NN);
    };

    // layer 0
    genconv(hbuf, 0, nullptr, hbuf);

    // layers 1..L-1
    for (int l = 1; l < LL; ++l) {
        const float* ng  = norm_g  + (size_t)(l - 1) * HH;
        const float* nb  = norm_b  + (size_t)(l - 1) * HH;
        const float* w1  = vn_W1   + (size_t)(l - 1) * HH * HH;
        const float* b1  = vn_b1   + (size_t)(l - 1) * HH;
        const float* vg  = vn_g    + (size_t)(l - 1) * HH;
        const float* vbt = vn_beta + (size_t)(l - 1) * HH;
        const float* w2  = vn_W2   + (size_t)(l - 1) * HH * HH;
        const float* b2  = vn_b2   + (size_t)(l - 1) * HH;

        ln_kernel<<<lnBlkN, T, 0, stream>>>(hbuf, h2, ng, nb, NN, 1);          // h2 = relu(LN(h))
        copy_f32_kernel<<<ghBlk, T, 0, stream>>>(vnst, vntmp, GG * HH);        // vntmp = vn
        vn_scatter_kernel<<<nodeBlk, T, 0, stream>>>(h2, batch, vntmp);        // += seg_sum(h2)
        vn_gemm_kernel<<<dim3(GG, HH / 128), 128, 0, stream>>>(vntmp, w1, b1, vnhid);
        ln_kernel<<<lnBlkG, T, 0, stream>>>(vnhid, vnhid, vg, vbt, GG, 1);     // relu(LN(...))
        vn_gemm_kernel<<<dim3(GG, HH / 128), 128, 0, stream>>>(vnhid, w2, b2, vnst);
        add_vn_kernel<<<nodeBlk, T, 0, stream>>>(h2, vnst, batch);             // h2 += vn[batch]
        genconv(h2, l, hbuf, hbuf);                                            // h = conv(h2)+h
    }

    // final LayerNorm -> output
    ln_kernel<<<lnBlkN, T, 0, stream>>>(hbuf, (float*)d_out,
                                        norm_g + (size_t)(LL - 1) * HH,
                                        norm_b + (size_t)(LL - 1) * HH, NN, 0);
}
